// MultiHeadAttention_27659589386802
// MI455X (gfx1250) — compile-verified, matrix operand order checked by reading, not run on hardware
//
#include <hip/hip_runtime.h>
#include <cstdint>

// ---- problem constants (match reference) ----
constexpr int Bc  = 2;
constexpr int Sc  = 2048;
constexpr int Dc  = 1024;
constexpr int Hc  = 16;
constexpr int DHc = 64;

typedef __attribute__((ext_vector_type(16))) __bf16 v16bf;
typedef __attribute__((ext_vector_type(8)))  __bf16 v8bf;
typedef __attribute__((ext_vector_type(4)))  __bf16 v4bf;
typedef __attribute__((ext_vector_type(8)))  float  v8f;

// TDM descriptor vector types (per CDNA5_HIP.md probe)
typedef __attribute__((ext_vector_type(4))) unsigned int tdm_v4u;
typedef __attribute__((ext_vector_type(8))) int          tdm_v8i;
typedef __attribute__((ext_vector_type(4))) int          tdm_v4i;

// Load a 16-element bf16 WMMA A/B fragment for one lane.
// CDNA5 16-bit operand layout (16x32): lanes 0-15 hold K {k..k+7, k+16..k+23},
// lanes 16-31 hold K {k+8..k+15, k+24..k+31} of their row/column.
__device__ __forceinline__ v16bf load_frag(const __bf16* __restrict__ row, int k, int hi) {
    const int off0 = hi ? 8 : 0;
    v8bf x = *(const v8bf*)(row + k + off0);
    v8bf y = *(const v8bf*)(row + k + off0 + 16);
    v16bf r;
#pragma unroll
    for (int i = 0; i < 8; ++i) { r[i] = x[i]; r[i + 8] = y[i]; }
    return r;
}

__device__ __forceinline__ v8f wmma_bf16(v16bf a, v16bf b, v8f c) {
    return __builtin_amdgcn_wmma_f32_16x16x32_bf16(false, a, false, b, (short)0, c, false, false);
}

// Issue a TDM load of a 16-row x 64-col bf16 tile (row stride 1024 elements)
// from global memory into LDS at byte offset lds_off. Descriptor layout per
// CDNA5 ISA (D# group 0/1; groups 2/3 zero for 2D tiles).
__device__ __forceinline__ void tdm_load_ktile(const __bf16* gsrc, unsigned lds_off) {
    unsigned long long ga = (unsigned long long)(uintptr_t)gsrc;
    tdm_v4u g0 = {};
    g0[0] = 1u;                                          // count=1, user mode
    g0[1] = lds_off;                                     // lds_addr (bytes)
    g0[2] = (unsigned)(ga & 0xffffffffu);                // global_addr[31:0]
    g0[3] = (unsigned)((ga >> 32) & 0x01ffffffu)         // global_addr[56:32]
          | (2u << 30);                                  // type = 2 ("image")
    tdm_v8i g1 = {};
    g1[0] = (int)(1u << 16);                             // data_size=1 -> 2 bytes
    g1[1] = (int)(64u << 16);                            // tensor_dim0 = 64
    g1[2] = (int)(16u << 16);                            // tensor_dim1 = 16
    g1[3] = (int)(64u << 16);                            // tile_dim0 = 64
    g1[4] = 16;                                          // tile_dim1 = 16
    g1[5] = 1024;                                        // tensor_dim0_stride = 1024
    tdm_v4i gz = {};
#if __clang_major__ >= 23
    tdm_v8i gz8 = {};
    __builtin_amdgcn_tensor_load_to_lds(g0, g1, gz, gz, gz8, 0);
#else
    __builtin_amdgcn_tensor_load_to_lds(g0, g1, gz, gz, 0);
#endif
}

// ---------------- f32 -> bf16 conversion (vectorized) ----------------
__global__ __launch_bounds__(256) void cvt_f32_bf16(const float4* __restrict__ src,
                                                    v4bf* __restrict__ dst, int n4) {
    int i = blockIdx.x * 256 + threadIdx.x;
    if (i < n4) {
        float4 f = src[i];
        v4bf o;
        o[0] = (__bf16)f.x; o[1] = (__bf16)f.y; o[2] = (__bf16)f.z; o[3] = (__bf16)f.w;
        dst[i] = o;
    }
}

// ---------------- GEMM: C = X @ W^T + bias ----------------
// mode 0: store bf16 [M,1024]
// mode 1: store bf16 per-head-transposed V^T: [B,H,DH,S]
// mode 2: store f32  [M,1024]
__global__ __launch_bounds__(256) void gemm_xwt(const __bf16* __restrict__ X,
                                                const __bf16* __restrict__ W,
                                                const float* __restrict__ bias,
                                                void* __restrict__ out,
                                                int mode) {
    const int tid  = threadIdx.x;
    const int lane = tid & 31;
    const int wave = tid >> 5;          // 8 waves
    const int wm   = wave >> 1;         // 0..3
    const int wn   = wave & 1;          // 0..1
    const int lr   = lane & 15;
    const int hi   = lane >> 4;

    const int m0 = blockIdx.x * 64 + wm * 16;
    const int n0 = blockIdx.y * 64 + wn * 32;

    const __bf16* Arow  = X + (size_t)(m0 + lr) * Dc;
    const __bf16* Brow0 = W + (size_t)(n0 + lr) * Dc;
    const __bf16* Brow1 = W + (size_t)(n0 + 16 + lr) * Dc;

    v8f acc0 = {}; v8f acc1 = {};
#pragma unroll 4
    for (int k = 0; k < Dc; k += 32) {
        v16bf a  = load_frag(Arow,  k, hi);
        v16bf b0 = load_frag(Brow0, k, hi);
        v16bf b1 = load_frag(Brow1, k, hi);
        acc0 = wmma_bf16(a, b0, acc0);
        acc1 = wmma_bf16(a, b1, acc1);
    }

    const float bias0 = bias[n0 + lr];
    const float bias1 = bias[n0 + 16 + lr];

#pragma unroll
    for (int r = 0; r < 8; ++r) {
        const int m  = m0 + r + hi * 8;
        const int c0 = n0 + lr;
        const int c1 = c0 + 16;
        const float v0 = acc0[r] + bias0;
        const float v1 = acc1[r] + bias1;
        if (mode == 2) {
            float* o = (float*)out;
            o[(size_t)m * Dc + c0] = v0;
            o[(size_t)m * Dc + c1] = v1;
        } else if (mode == 1) {
            __bf16* o = (__bf16*)out;
            const int bb = m >> 11, ss = m & (Sc - 1);
            o[(((size_t)bb * Hc + (c0 >> 6)) * DHc + (c0 & 63)) * Sc + ss] = (__bf16)v0;
            o[(((size_t)bb * Hc + (c1 >> 6)) * DHc + (c1 & 63)) * Sc + ss] = (__bf16)v1;
        } else {
            __bf16* o = (__bf16*)out;
            o[(size_t)m * Dc + c0] = (__bf16)v0;
            o[(size_t)m * Dc + c1] = (__bf16)v1;
        }
    }
}

// ---------------- fused attention ----------------
// One block = one (b, h, 16-query-row tile). 256 threads = 8 waves.
// Dynamic LDS layout (bytes):
//   [0,     128K)  sc    : 16 x 2048 f32 scores / exp values
//   [128K,  192K)  wbuf  : 16 x 2048 bf16 softmax weights (A-operand for P@V)
//   [192K,  200K)  pbuf  : 2 x 4 x 16 x 16 f32 split-K partials for phase 3
//   [200K,  232K)  kstage: 8 waves x 2 slots x 2KB TDM-staged K tiles
constexpr unsigned SC_OFF   = 0;
constexpr unsigned WBUF_OFF = 16 * Sc * 4;                 // 131072
constexpr unsigned PBUF_OFF = WBUF_OFF + 16 * Sc * 2;      // 196608
constexpr unsigned KST_OFF  = PBUF_OFF + 2 * 4 * 16 * 16 * 4;  // 204800
constexpr unsigned LDS_TOTAL = KST_OFF + 8 * 2 * 2048;     // 237568

__global__ __launch_bounds__(256) void attn_fused(const __bf16* __restrict__ Qp,
                                                  const __bf16* __restrict__ Kp,
                                                  const __bf16* __restrict__ Vt,
                                                  float* __restrict__ wout,
                                                  __bf16* __restrict__ attnB) {
    extern __shared__ char smem[];
    float*  sc   = (float*)(smem + SC_OFF);
    __bf16* wbuf = (__bf16*)(smem + WBUF_OFF);
    float*  pbuf = (float*)(smem + PBUF_OFF);

    const int tid  = threadIdx.x;
    const int lane = tid & 31;
    const int wave = tid >> 5;          // 0..7
    const int lr   = lane & 15;
    const int hi   = lane >> 4;

    const int t  = blockIdx.x;
    const int qt = t & (Sc / 16 - 1);   // 0..127
    const int h  = (t >> 7) & (Hc - 1);
    const int b  = t >> 11;
    const int q0 = qt * 16;

    // ---- phase 1: scores = Q K^T / sqrt(DH); K tiles staged via TDM ----
    const __bf16* Qrow = Qp + (size_t)(b * Sc + q0 + lr) * Dc + h * DHc;
    const v16bf aQ0 = load_frag(Qrow, 0,  hi);
    const v16bf aQ1 = load_frag(Qrow, 32, hi);

    const __bf16* kbase = Kp + (size_t)b * Sc * Dc + h * DHc;
    const unsigned slot0 = KST_OFF + (unsigned)wave * 4096;
    const int nblocks = qt + 1;         // causal: key blocks up to the diagonal

    if (wave < nblocks) tdm_load_ktile(kbase + (size_t)wave * 16 * Dc, slot0);
    int it = 0;
    for (int kb = wave; kb < nblocks; kb += 8, ++it) {
        const int nxt = kb + 8;
        if (nxt < nblocks) {
            tdm_load_ktile(kbase + (size_t)nxt * 16 * Dc, slot0 + (((it + 1) & 1) << 11));
            __builtin_amdgcn_s_wait_tensorcnt(1);   // current tile complete
        } else {
            __builtin_amdgcn_s_wait_tensorcnt(0);
        }
        const __bf16* kt = (const __bf16*)(smem + slot0 + ((it & 1) << 11)) + lr * DHc;
        v16bf bK0 = load_frag(kt, 0,  hi);          // ds_load_b128 from staged tile
        v16bf bK1 = load_frag(kt, 32, hi);
        v8f acc = {};
        acc = wmma_bf16(aQ0, bK0, acc);
        acc = wmma_bf16(aQ1, bK1, acc);
#pragma unroll
        for (int r = 0; r < 8; ++r) {
            const int m    = r + hi * 8;
            const int kidx = kb * 16 + lr;
            sc[m * Sc + kidx] = (kidx <= q0 + m) ? acc[r] * 0.125f : -1.0e10f;
        }
    }
    __syncthreads();

    // ---- phase 2: row softmax; f32 weights to global, bf16 to LDS ----
    const int scored = nblocks * 16;
    const int kend   = (scored + 31) & ~31;
#pragma unroll
    for (int j = 0; j < 2; ++j) {
        const int row   = wave * 2 + j;
        const int qidx  = q0 + row;
        const int limit = qidx + 1;
        float* srow = sc + row * Sc;
        float mx = -3.0e38f;
        for (int c = lane; c < limit; c += 32) mx = fmaxf(mx, srow[c]);
#pragma unroll
        for (int off = 16; off; off >>= 1) mx = fmaxf(mx, __shfl_xor(mx, off, 32));
        float sum = 0.f;
        for (int c = lane; c < limit; c += 32) {
            const float e = __expf(srow[c] - mx);
            srow[c] = e;                             // reuse: single exp per element
            sum += e;
        }
#pragma unroll
        for (int off = 16; off; off >>= 1) sum += __shfl_xor(sum, off, 32);
        const float inv = 1.0f / sum;
        float* wrow = wout + ((size_t)(b * Hc + h) * Sc + qidx) * Sc;
        for (int c = lane * 4; c < Sc; c += 128) {   // float4 -> global_store_b128
            float4 w;
            w.x = (c + 0 < limit) ? srow[c + 0] * inv : 0.f;
            w.y = (c + 1 < limit) ? srow[c + 1] * inv : 0.f;
            w.z = (c + 2 < limit) ? srow[c + 2] * inv : 0.f;
            w.w = (c + 3 < limit) ? srow[c + 3] * inv : 0.f;
            *(float4*)(wrow + c) = w;
            if (c < kend) {
                v4bf wb;
                wb[0] = (__bf16)w.x; wb[1] = (__bf16)w.y;
                wb[2] = (__bf16)w.z; wb[3] = (__bf16)w.w;
                *(v4bf*)(wbuf + row * Sc + c) = wb;
            }
        }
    }
    __syncthreads();

    // ---- phase 3: attn = weights @ V, split-K across wave pairs ----
    const int dt = wave & 3;            // d-tile (16 cols of DH)
    const int hf = wave >> 2;           // K-range half
    const int nchunks = kend >> 5;
    const int kh = (((nchunks + 1) >> 1) << 5);      // 32-aligned split point
    const int kbeg = hf ? kh : 0;
    const int kfin = hf ? kend : kh;

    const __bf16* Vrow = Vt + ((size_t)(b * Hc + h) * DHc + dt * 16 + lr) * Sc;
    const __bf16* Wrow = wbuf + (size_t)lr * Sc;
    v8f acc = {};
    for (int k = kbeg; k < kfin; k += 32) {
        v16bf a = load_frag(Wrow, k, hi);
        v16bf v = load_frag(Vrow, k, hi);
        acc = wmma_bf16(a, v, acc);
    }
#pragma unroll
    for (int r = 0; r < 8; ++r) {
        const int m = r + hi * 8;
        pbuf[(((hf << 2) | dt) * 16 + m) * 16 + lr] = acc[r];
    }
    __syncthreads();

    // combine halves and emit bf16 attn rows
#pragma unroll
    for (int i = 0; i < 4; ++i) {
        const int e   = tid + (i << 8);              // 0..1023
        const int row = e >> 6;
        const int col = e & 63;
        const int dt2 = col >> 4, n = col & 15;
        const float v0 = pbuf[(dt2 * 16 + row) * 16 + n];
        const float v1 = pbuf[((4 + dt2) * 16 + row) * 16 + n];
        attnB[(size_t)(b * Sc + q0 + row) * Dc + h * DHc + col] = (__bf16)(v0 + v1);
    }
}

// ---------------- host launcher ----------------
extern "C" void kernel_launch(void* const* d_in, const int* in_sizes, int n_in,
                              void* d_out, int out_size, void* d_ws, size_t ws_size,
                              hipStream_t stream) {
    (void)in_sizes; (void)n_in; (void)out_size; (void)ws_size;

    const float* q  = (const float*)d_in[0];
    const float* k  = (const float*)d_in[1];
    const float* v  = (const float*)d_in[2];
    // d_in[3] = causal mask (known statically) -- unused
    const float* Wq = (const float*)d_in[4];
    const float* bq = (const float*)d_in[5];
    const float* Wk = (const float*)d_in[6];
    const float* bk = (const float*)d_in[7];
    const float* Wv = (const float*)d_in[8];
    const float* bv = (const float*)d_in[9];
    const float* Wo = (const float*)d_in[10];
    const float* bo = (const float*)d_in[11];

    float* xout = (float*)d_out;
    float* wout = xout + (size_t)Bc * Sc * Dc;

    const size_t NQ = (size_t)Bc * Sc * Dc;        // 4,194,304 elements
    const size_t NW = (size_t)Dc * Dc;             // 1,048,576 elements

    char* ws = (char*)d_ws;
    __bf16* qb    = (__bf16*)ws;  ws += NQ * 2;
    __bf16* kb    = (__bf16*)ws;  ws += NQ * 2;
    __bf16* vb    = (__bf16*)ws;  ws += NQ * 2;
    __bf16* wqb   = (__bf16*)ws;  ws += NW * 2;
    __bf16* wkb   = (__bf16*)ws;  ws += NW * 2;
    __bf16* wvb   = (__bf16*)ws;  ws += NW * 2;
    __bf16* wob   = (__bf16*)ws;  ws += NW * 2;
    __bf16* Qp    = (__bf16*)ws;  ws += NQ * 2;
    __bf16* Kp    = (__bf16*)ws;  ws += NQ * 2;
    __bf16* Vt    = (__bf16*)ws;  ws += NQ * 2;
    __bf16* attnB = (__bf16*)ws;  ws += NQ * 2;

    // 1) convert inputs and weight matrices to bf16 (float4 vectorized)
    cvt_f32_bf16<<<(int)(NQ / 1024), 256, 0, stream>>>((const float4*)q,  (v4bf*)qb,  (int)(NQ / 4));
    cvt_f32_bf16<<<(int)(NQ / 1024), 256, 0, stream>>>((const float4*)k,  (v4bf*)kb,  (int)(NQ / 4));
    cvt_f32_bf16<<<(int)(NQ / 1024), 256, 0, stream>>>((const float4*)v,  (v4bf*)vb,  (int)(NQ / 4));
    cvt_f32_bf16<<<(int)(NW / 1024), 256, 0, stream>>>((const float4*)Wq, (v4bf*)wqb, (int)(NW / 4));
    cvt_f32_bf16<<<(int)(NW / 1024), 256, 0, stream>>>((const float4*)Wk, (v4bf*)wkb, (int)(NW / 4));
    cvt_f32_bf16<<<(int)(NW / 1024), 256, 0, stream>>>((const float4*)Wv, (v4bf*)wvb, (int)(NW / 4));
    cvt_f32_bf16<<<(int)(NW / 1024), 256, 0, stream>>>((const float4*)Wo, (v4bf*)wob, (int)(NW / 4));

    // 2) Q/K/V projections (WMMA GEMMs); V stored per-head transposed
    dim3 gg((Bc * Sc) / 64, Dc / 64);
    gemm_xwt<<<gg, 256, 0, stream>>>(qb, wqb, bq, (void*)Qp, 0);
    gemm_xwt<<<gg, 256, 0, stream>>>(kb, wkb, bk, (void*)Kp, 0);
    gemm_xwt<<<gg, 256, 0, stream>>>(vb, wvb, bv, (void*)Vt, 1);

    // 3) fused scores (TDM-staged K) + softmax + weights-out + P@V
    const int attn_blocks = Bc * Hc * (Sc / 16);             // 4096
    attn_fused<<<attn_blocks, 256, LDS_TOTAL, stream>>>(Qp, Kp, Vt, wout, attnB);

    // 4) output projection -> f32 x
    gemm_xwt<<<gg, 256, 0, stream>>>(attnB, wob, bo, (void*)xout, 2);
}
